// DotProductAttention_29944511988402
// MI455X (gfx1250) — matrix operand order repaired and checked by reading, not compiled, not run on hardware
//
#include <hip/hip_runtime.h>

typedef _Float16 v16h __attribute__((ext_vector_type(16)));
typedef _Float16 h2   __attribute__((ext_vector_type(2)));
typedef float    v8f  __attribute__((ext_vector_type(8)));

#define B_  2
#define H_  16
#define L_  2048
#define D_  128
#define BLOCK_N 32
#define NEGV (-1e9f)
#define KPAD 132   // 132 floats = 528B rows: 16B-aligned, banks advance 4/row

// v_cvt_pk_f16_f32: pack two fp32 into two f16 (builtin returns __fp16x2;
// bit-identical to _Float16x2, so bit_cast).
__device__ __forceinline__ h2 pk(float a, float b) {
  auto t = __builtin_amdgcn_cvt_pkrtz(a, b);
  return __builtin_bit_cast(h2, t);
}

// Issue one 32x128 fp32 K tile + V tile as async global->LDS b128 copies.
// 8 + 8 instructions per thread (16 per wave-slot => ASYNCcnt +16 per wave).
__device__ __forceinline__ void issue_tile_async(
    const float* __restrict__ gK, const float* __restrict__ gV,
    int kbase, int tid, float (*kl)[KPAD], float (*vl)[KPAD]) {
  #pragma unroll
  for (int i = 0; i < 8; ++i) {
    const int flat = i * 128 + tid;
    const int row  = flat >> 5;          // key row within tile
    const int d0   = (flat & 31) << 2;   // feature dim (float4 granule)
    // generic pointer low 32 bits == LDS byte offset (ISA aperture rule)
    const unsigned lk = (unsigned)(uintptr_t)&kl[row][d0];
    const unsigned lv = (unsigned)(uintptr_t)&vl[row][d0];
    const unsigned go = (unsigned)(((kbase + row) * D_ + d0) * sizeof(float));
    asm volatile("global_load_async_to_lds_b128 %0, %1, %2 offset:0"
                 :: "v"(lk), "v"(go), "s"(gK) : "memory");
    asm volatile("global_load_async_to_lds_b128 %0, %1, %2 offset:0"
                 :: "v"(lv), "v"(go), "s"(gV) : "memory");
  }
}

__global__ __launch_bounds__(128)
void fa_fwd_kernel(const float* __restrict__ Q, const float* __restrict__ K,
                   const float* __restrict__ V, const int* __restrict__ valid_lens,
                   float* __restrict__ O) {
  __shared__ __align__(16) float kbuf[2][BLOCK_N][KPAD];
  __shared__ __align__(16) float vbuf[2][BLOCK_N][KPAD];

  const int tid  = threadIdx.x;
  const int wave = tid >> 5;
  const int lane = tid & 31;
  const int half = lane >> 4;
  const int ln   = lane & 15;

  const int bh    = blockIdx.y;               // b*H + h
  const int b     = bh / H_;
  const int qbase = blockIdx.x * 64 + wave * 16;

  const size_t head_off = (size_t)bh * L_ * D_;
  const float* Qh = Q + head_off;
  const float* Kh = K + head_off;
  const float* Vh = V + head_off;
  float*       Oh = O + head_off;

  const int vlen = valid_lens[b];
  const int nkb  = (vlen + BLOCK_N - 1) / BLOCK_N;   // uniform across block

  const float scale = 0.088388347648318447f;  // 1/sqrt(128)

  // ---- Q^T B-fragments (S^T = K * Q^T): lane = query column (N),
  // elements = d within 32-chunk (K dim, table order). Loaded once, reused every iteration.
  v16h qb[4];
  #pragma unroll
  for (int c = 0; c < 4; ++c) {
    #pragma unroll
    for (int e = 0; e < 16; ++e)
      qb[c][e] = (_Float16)(Qh[(size_t)(qbase + ln) * D_ + c * 32 + 8 * half + (e & 7) + ((e >> 3) << 4)] * scale);
  }

  v8f zero = {};
  v8f o[8];
  #pragma unroll
  for (int t = 0; t < 8; ++t) o[t] = zero;
  float m_q = -INFINITY;   // per-lane: softmax max of query column `ln`
  float l_q = 0.0f;        // per-lane: softmax denom of query column `ln`

  issue_tile_async(Kh, Vh, 0, tid, kbuf[0], vbuf[0]);

  for (int kb = 0; kb < nkb; ++kb) {
    const int cur   = kb & 1;
    const int kbase = kb * BLOCK_N;

    if (kb + 1 < nkb) {   // prefetch next tile into the other buffer
      issue_tile_async(Kh, Vh, kbase + BLOCK_N, tid,
                       kbuf[cur ^ 1], vbuf[cur ^ 1]);
      asm volatile("s_wait_asynccnt 0x10" ::: "memory"); // tile kb landed
    } else {
      asm volatile("s_wait_asynccnt 0x0" ::: "memory");
    }
    __syncthreads();  // all waves' async copies for tile kb visible

    // ---- S^T = K * Q^T. A = K rows (lane = key, elems = d per 7.12.2 perm:
    // e<8 -> d = 8*half + e, e>=8 -> d = 16 + 8*half + (e-8)) => two
    // contiguous 8-float runs per 32-chunk: pure ds_load_b128 + cvt_pk.
    v8f s[2];
    #pragma unroll
    for (int g = 0; g < 2; ++g) {
      const float* krow = &kbuf[cur][g * 16 + ln][0];
      v8f acc = zero;
      #pragma unroll
      for (int c = 0; c < 4; ++c) {
        const float* kp = krow + c * 32 + 8 * half;
        const float4 a0 = *(const float4*)(kp);
        const float4 a1 = *(const float4*)(kp + 4);
        const float4 b0 = *(const float4*)(kp + 16);
        const float4 b1 = *(const float4*)(kp + 20);
        v16h kf;
        h2 t;
        t = pk(a0.x, a0.y); kf[0]  = t.x; kf[1]  = t.y;
        t = pk(a0.z, a0.w); kf[2]  = t.x; kf[3]  = t.y;
        t = pk(a1.x, a1.y); kf[4]  = t.x; kf[5]  = t.y;
        t = pk(a1.z, a1.w); kf[6]  = t.x; kf[7]  = t.y;
        t = pk(b0.x, b0.y); kf[8]  = t.x; kf[9]  = t.y;
        t = pk(b0.z, b0.w); kf[10] = t.x; kf[11] = t.y;
        t = pk(b1.x, b1.y); kf[12] = t.x; kf[13] = t.y;
        t = pk(b1.z, b1.w); kf[14] = t.x; kf[15] = t.y;
        acc = __builtin_amdgcn_wmma_f32_16x16x32_f16(
            false, kf, false, qb[c], (short)0, acc, false, false);
      }
      s[g] = acc;
    }

    // ---- mask: element r of group g is key kbase + g*16 + r + 8*half
    const int key0 = kbase + 8 * half;
    #pragma unroll
    for (int g = 0; g < 2; ++g)
      #pragma unroll
      for (int r = 0; r < 8; ++r)
        s[g][r] = (key0 + g * 16 + r < vlen) ? s[g][r] : NEGV;

    // ---- online softmax: all 32 keys of this block for query `ln` live in
    // this lane (16 values) + partner lane (xor 16). No bpermute chains.
    float mij = s[0][0];
    #pragma unroll
    for (int g = 0; g < 2; ++g)
      #pragma unroll
      for (int r = 0; r < 8; ++r)
        if (g | r) mij = fmaxf(mij, s[g][r]);
    mij = fmaxf(mij, __shfl_xor(mij, 16, 32));
    const float mnew = fmaxf(m_q, mij);
    const float aq   = __expf(m_q - mnew);
    m_q = mnew;

    float p[2][8];
    float rs = 0.0f;
    #pragma unroll
    for (int g = 0; g < 2; ++g)
      #pragma unroll
      for (int r = 0; r < 8; ++r) {
        p[g][r] = __expf(s[g][r] - mnew);
        rs += p[g][r];
      }
    rs += __shfl_xor(rs, 16, 32);
    l_q = l_q * aq + rs;

    // ---- broadcast alpha (query-indexed) to O layout (elem r = query r+8h)
    float abc[8];
    #pragma unroll
    for (int r = 0; r < 8; ++r) abc[r] = __shfl(aq, r + 8 * half, 32);
    #pragma unroll
    for (int t = 0; t < 8; ++t)
      #pragma unroll
      for (int r = 0; r < 8; ++r) o[t][r] *= abc[r];

    // ---- P is already in A layout in-lane: af[e] = p[e>>3][e&7]
    v16h af;
    #pragma unroll
    for (int g = 0; g < 2; ++g)
      #pragma unroll
      for (int j = 0; j < 4; ++j) {
        const h2 t = pk(p[g][2 * j], p[g][2 * j + 1]);
        af[g * 8 + 2 * j]     = t.x;
        af[g * 8 + 2 * j + 1] = t.y;
      }

    // ---- O += P * V. B = V tile: lane = d col (N), elems = key (K dim, table order).
    #pragma unroll
    for (int t = 0; t < 8; ++t) {
      v16h vf;
      #pragma unroll
      for (int e = 0; e < 16; ++e)
        vf[e] = (_Float16)vbuf[cur][8 * half + (e & 7) + ((e >> 3) << 4)][t * 16 + ln];
      o[t] = __builtin_amdgcn_wmma_f32_16x16x32_f16(
          false, af, false, vf, (short)0, o[t], false, false);
    }

    __syncthreads();  // tile-kb LDS reads done before next issue overwrites
  }

  // ---- epilogue: fetch denominators into O layout, normalize, store fp32
  float linv[8];
  #pragma unroll
  for (int r = 0; r < 8; ++r)
    linv[r] = 1.0f / __shfl(l_q, r + 8 * half, 32);
  #pragma unroll
  for (int r = 0; r < 8; ++r) {
    float* orow = Oh + (size_t)(qbase + r + 8 * half) * D_;
    #pragma unroll
    for (int t = 0; t < 8; ++t)
      orow[t * 16 + ln] = o[t][r] * linv[r];
  }
}

extern "C" void kernel_launch(void* const* d_in, const int* in_sizes, int n_in,
                              void* d_out, int out_size, void* d_ws, size_t ws_size,
                              hipStream_t stream) {
  const float* q  = (const float*)d_in[0];
  const float* k  = (const float*)d_in[1];
  const float* v  = (const float*)d_in[2];
  const int*   vl = (const int*)d_in[3];
  float* out = (float*)d_out;

  dim3 grid(L_ / 64, B_ * H_);   // (query tiles of 64, B*H)
  fa_fwd_kernel<<<grid, 128, 0, stream>>>(q, k, v, vl, out);
}
